// KANConvolution_7413113553459
// MI455X (gfx1250) — matrix operand order repaired and checked by reading, not compiled, not run on hardware
//
#include <hip/hip_runtime.h>

typedef __attribute__((ext_vector_type(16))) _Float16 v16h;
typedef __attribute__((ext_vector_type(8)))  _Float16 v8h;
typedef __attribute__((ext_vector_type(8)))  float    v8f;
typedef __attribute__((ext_vector_type(4)))  unsigned v4u;
typedef __attribute__((ext_vector_type(8)))  unsigned v8u;
typedef _Float16 half_t;

// Problem constants (from reference): B=8,H=32,W=32,C=64,F=64,KH=KW=3,G=16
#define NPIX   8192           // B*H*W
#define KB1    288            // spline K-blocks: 2 taps * 16 grid = 32 K per block
#define KB2    18             // silu  K-blocks: 32 taps per block (18*32 = 576)
#define SLICE_HALVES 2048     // one K-block's B fragments: 64 f * 32 halves = 4KB
#define WSB_HALVES  (KB1 * SLICE_HALVES)   // 589824 halves of swizzled wcp
#define WSW2_HALVES (KB2 * SLICE_HALVES)   // 36864 halves of swizzled w2

// Workspace slice layout (halves), chosen for conflict-light LDS reads:
//   [kb][hl(2)][q(128)][e(8)]   with q = (nb*2 + khalf)*16 + lane16
//   hl=0 -> fragment halves 0..7, hl=1 -> fragment halves 8..15
// Reader lane (lane = khalf*16+lane16) of n-block nb reads chunk q = nb*32+lane:
// 16B at q*16 and 16B at 2048+q*16 -> 32 lanes hit 32 consecutive 16B chunks.

__device__ __forceinline__ void decode_q(int q, int hl, int e,
                                         int& f, int& klocal) {
    int nb     = q >> 5;
    int khalf  = (q >> 4) & 1;
    int lane16 = q & 15;
    f      = nb * 16 + lane16;
    klocal = khalf * 8 + hl * 16 + e;
}

// ---------------- prep: swizzle w1*control_points into staged-B layout ----------
__global__ __launch_bounds__(256)
void kan_prep_wcp(const float* __restrict__ cp, const float* __restrict__ w1,
                  half_t* __restrict__ wsB) {
    int idx = blockIdx.x * 256 + threadIdx.x;
    if (idx >= WSB_HALVES) return;
    int e  = idx & 7;
    int q  = (idx >> 3) & 127;
    int hl = (idx >> 10) & 1;
    int kb = idx >> 11;
    int f, klocal;
    decode_q(q, hl, e, f, klocal);
    int t = kb * 2 + (klocal >> 4);        // tap 0..575, order c*9 + i*3 + j
    int g = klocal & 15;
    int c = t / 9;
    int r = t - 9 * c;
    int i = r / 3;
    int j = r - 3 * i;
    int wi   = ((f * 64 + c) * 3 + i) * 3 + j;     // w1 index (F,C,3,3)
    wsB[idx] = (half_t)(w1[wi] * cp[wi * 16 + g]); // cp index (F,C,3,3,16)
}

// ---------------- prep: swizzle w2 into staged-B layout -------------------------
__global__ __launch_bounds__(256)
void kan_prep_w2(const float* __restrict__ w2, half_t* __restrict__ wsW2) {
    int idx = blockIdx.x * 256 + threadIdx.x;
    if (idx >= WSW2_HALVES) return;
    int e  = idx & 7;
    int q  = (idx >> 3) & 127;
    int hl = (idx >> 10) & 1;
    int kb = idx >> 11;
    int f, klocal;
    decode_q(q, hl, e, f, klocal);
    int t = kb * 32 + klocal;                      // 0..575 exactly
    int c = t / 9;
    int r = t - 9 * c;
    int i = r / 3;
    int j = r - 3 * i;
    wsW2[idx] = (half_t)w2[((f * 64 + c) * 3 + i) * 3 + j];
}

// ---------------- gfx1250 TDM: one instruction moves a whole 4KB slice ----------
// D# group0: count=1 | lds_addr | global_addr[56:0] | type=2
// D# group1: data_size=8B, 1-D tile: tensor_dim0=tile_dim0=stride=512 units,
//            tensor_dim1=1, no multicast/pad/iterate/atomic-barrier.
__device__ __forceinline__ void tdm_load_slice(const void* gsrc,
                                               unsigned lds_addr) {
    unsigned long long ga = (unsigned long long)(uintptr_t)gsrc;
    v4u g0;
    g0.x = 1u;                               // count=1, user descriptor
    g0.y = lds_addr;                         // LDS byte address
    g0.z = (unsigned)ga;                     // global_addr[31:0]
    g0.w = (unsigned)(ga >> 32) | (2u << 30);// global_addr[56:32], type=2
    v8u g1;
    g1.s0 = 3u << 16;                        // data_size = 3 (8-byte units)
    g1.s1 = 512u << 16;                      // tensor_dim0[15:0] in bits 63:48
    g1.s2 = 1u << 16;                        // tensor_dim0[31:16]=0, tensor_dim1=1
    g1.s3 = 512u << 16;                      // tile_dim0 = 512 (bits 127:112)
    g1.s4 = 0u;                              // tile_dim1 = tile_dim2 = 0 (unused)
    g1.s5 = 512u;                            // tensor_dim0_stride low32
    g1.s6 = 0u;
    g1.s7 = 0u;
    asm volatile("tensor_load_to_lds %0, %1" :: "s"(g0), "s"(g1) : "memory");
}

#if __has_builtin(__builtin_amdgcn_s_wait_tensorcnt)
#define WAIT_TENSOR(n) __builtin_amdgcn_s_wait_tensorcnt(n)
#else
#define WAIT_TENSOR(n) asm volatile("s_wait_tensorcnt %0" :: "i"(n) : "memory")
#endif

// ---------------- gfx1250 async-to-LDS (per-lane) path --------------------------
__device__ __forceinline__ void async_copy16(unsigned lds_off, const void* gsrc) {
    asm volatile("global_load_async_to_lds_b128 %0, %1, off"
                 :: "v"(lds_off), "v"((unsigned long long)(uintptr_t)gsrc)
                 : "memory");
}

#if __has_builtin(__builtin_amdgcn_s_wait_asynccnt)
#define WAIT_ASYNC(n) __builtin_amdgcn_s_wait_asynccnt(n)
#else
#define WAIT_ASYNC(n) asm volatile("s_wait_asynccnt %0" :: "i"(n) : "memory")
#endif

// Each of the 128 threads copies 32B of the 4KB slice (two b128 async loads).
__device__ __forceinline__ void stage_slice_async(const half_t* __restrict__ g,
                                                  unsigned lds_base, int tid) {
    const char* s = (const char*)g;
    unsigned d = lds_base + (unsigned)(tid * 16);
    async_copy16(d,        s + tid * 16);
    async_copy16(d + 2048, s + 2048 + tid * 16);
}

// Gather one SAME-padded patch element for pixel (h,w) of image xb, tap t=c*9+i*3+j
__device__ __forceinline__ float patch_val(const float* __restrict__ xb,
                                           int h, int w, int t) {
    int c  = t / 9;
    int r  = t - 9 * c;
    int i  = r / 3;
    int j  = r - 3 * i;
    int hh = h + i - 1;
    int ww = w + j - 1;
    if ((unsigned)hh >= 32u || (unsigned)ww >= 32u) return 0.0f;
    return xb[(((hh << 5) + ww) << 6) + c];        // (H,W,C) inner layout
}

__device__ __forceinline__ float silu_f(float v) {
    return v / (1.0f + __expf(-v));
}

// ---------------- main kernel: 1 wave = 16 pixels x 64 filters ------------------
__global__ __launch_bounds__(128)
void kan_conv_wmma(const float* __restrict__ x,
                   const half_t* __restrict__ wsB,
                   const half_t* __restrict__ wsW2,
                   float* __restrict__ out) {
    __shared__ __align__(16) half_t smem[2][SLICE_HALVES];   // double buffer, 8KB

    const int tid    = threadIdx.x;
    const int wave   = tid >> 5;            // 4 waves per workgroup
    const int lane   = tid & 31;            // wave32
    const int lane16 = lane & 15;
    const int khalf  = lane >> 4;           // which K half this lane supplies
    const bool stager = (wave == 0);        // wave 0 drives the TDM
    const int P0     = blockIdx.x * 64 + wave * 16;    // tile base pixel (M dim)
    const int p      = P0 + lane16;                    // this lane's pixel row
    const int b      = p >> 10;
    const int h      = (p >> 5) & 31;
    const int w      = p & 31;
    const float* xb  = x + ((size_t)b << 16);          // b * 32*32*64
    const unsigned lds0 = (unsigned)(uintptr_t)&smem[0][0];
    const unsigned lds1 = (unsigned)(uintptr_t)&smem[1][0];

    v8f acc0 = {}, acc1 = {}, acc2 = {}, acc3 = {};

    // Assemble this lane's v16h B fragment from the staged slice in LDS.
    auto load_bf = [&](int cur, int nb) -> v16h {
        const half_t* lb = &smem[cur][0];
        const int q = nb * 32 + lane;
        v8h lo = *(const v8h*)(lb + (size_t)q * 8);
        v8h hi = *(const v8h*)(lb + 1024 + (size_t)q * 8);
        return __builtin_shufflevector(lo, hi, 0, 1, 2, 3, 4, 5, 6, 7,
                                       8, 9, 10, 11, 12, 13, 14, 15);
    };

    auto mma4 = [&](const v16h& a, int cur) {
        v16h bf;
        bf = load_bf(cur, 0);
        acc0 = __builtin_amdgcn_wmma_f32_16x16x32_f16(false, a, false, bf,
                                                      (short)0, acc0, false, false);
        bf = load_bf(cur, 1);
        acc1 = __builtin_amdgcn_wmma_f32_16x16x32_f16(false, a, false, bf,
                                                      (short)0, acc1, false, false);
        bf = load_bf(cur, 2);
        acc2 = __builtin_amdgcn_wmma_f32_16x16x32_f16(false, a, false, bf,
                                                      (short)0, acc2, false, false);
        bf = load_bf(cur, 3);
        acc3 = __builtin_amdgcn_wmma_f32_16x16x32_f16(false, a, false, bf,
                                                      (short)0, acc3, false, false);
    };

    // Phase 1 A-build: lanes 0-15 hold K 0-7 & 16-23, lanes 16-31 hold K 8-15 &
    // 24-31 (16-bit 16x32 A layout). K chunk [0,16)=tap t0 grid 0..15, [16,32)=t1.
    const int glo = khalf * 8;
    auto run_spline = [&](int kb, int cur) {
        const int t0 = kb * 2;
        const float pv0 = patch_val(xb, h, w, t0);
        const float pv1 = patch_val(xb, h, w, t0 + 1);
        float xc0 = fminf(fmaxf(pv0, -1.0f), 1.0f);
        float xc1 = fminf(fmaxf(pv1, -1.0f), 1.0f);
        float u0  = (xc0 + 1.0f) * 7.5f;    // linspace(-1,1,16): 1/h = 7.5
        float u1  = (xc1 + 1.0f) * 7.5f;
        int  i0   = (int)floorf(u0);  i0 = i0 < 0 ? 0 : (i0 > 14 ? 14 : i0);
        int  i1   = (int)floorf(u1);  i1 = i1 < 0 ? 0 : (i1 > 14 ? 14 : i1);
        float fr0 = fminf(fmaxf(u0 - (float)i0, 0.0f), 1.0f);
        float fr1 = fminf(fmaxf(u1 - (float)i1, 0.0f), 1.0f);
        float om0 = 1.0f - fr0;
        float om1 = 1.0f - fr1;
        v16h a;
#pragma unroll
        for (int e = 0; e < 8; ++e) {
            int g = glo + e;
            float v0 = (g == i0) ? om0 : ((g == i0 + 1) ? fr0 : 0.0f);
            float v1 = (g == i1) ? om1 : ((g == i1 + 1) ? fr1 : 0.0f);
            a[e]     = (half_t)v0;          // tap t0 -> K chunk 0
            a[8 + e] = (half_t)v1;          // tap t1 -> K chunk 1
        }
        mma4(a, cur);
    };

    auto run_silu = [&](int kb, int cur) {
        const int tb = kb * 32 + khalf * 8;
        v16h a;
#pragma unroll
        for (int e = 0; e < 8; ++e) {
            a[e]     = (half_t)silu_f(patch_val(xb, h, w, tb + e));      // K 0-15
            a[8 + e] = (half_t)silu_f(patch_val(xb, h, w, tb + 16 + e)); // K 16-31
        }
        mma4(a, cur);
    };

    // ===== phase 1: spline GEMM (K=9216), B staged by TDM, double-buffered ======
    if (stager) tdm_load_slice(wsB, lds0);
    for (int kb = 0; kb < KB1 - 1; ++kb) {
        const int cur = kb & 1;
        if (stager) {
            tdm_load_slice(wsB + (size_t)(kb + 1) * SLICE_HALVES,
                           cur ? lds0 : lds1);
            WAIT_TENSOR(1);     // slice kb landed; slice kb+1 still in flight
        }
        __syncthreads();
        run_spline(kb, cur);
        __syncthreads();        // slice consumed by all waves before reuse
    }
    if (stager) WAIT_TENSOR(0);
    __syncthreads();
    run_spline(KB1 - 1, (KB1 - 1) & 1);
    __syncthreads();

    // ===== phase 2: silu(patch) @ w2 (K=576), per-lane async-to-LDS staging =====
    stage_slice_async(wsW2, lds0, tid);
    for (int kb = 0; kb < KB2 - 1; ++kb) {
        const int cur = kb & 1;
        stage_slice_async(wsW2 + (size_t)(kb + 1) * SLICE_HALVES,
                          cur ? lds0 : lds1, tid);
        WAIT_ASYNC(2);
        __syncthreads();
        run_silu(kb, cur);
        __syncthreads();
    }
    WAIT_ASYNC(0);
    __syncthreads();
    run_silu(KB2 - 1, (KB2 - 1) & 1);

    // ===== store: C/D layout -> row = r + 8*khalf, col = lane16 (per N-block) ===
    float* op = out + (size_t)(P0 + khalf * 8) * 64 + lane16;
#pragma unroll
    for (int r = 0; r < 8; ++r) {
        op[(size_t)r * 64 +  0] = acc0[r];
        op[(size_t)r * 64 + 16] = acc1[r];
        op[(size_t)r * 64 + 32] = acc2[r];
        op[(size_t)r * 64 + 48] = acc3[r];
    }
}

extern "C" void kernel_launch(void* const* d_in, const int* in_sizes, int n_in,
                              void* d_out, int out_size, void* d_ws, size_t ws_size,
                              hipStream_t stream) {
    const float* x  = (const float*)d_in[0];   // (8,32,32,64)
    const float* cp = (const float*)d_in[1];   // (64,64,3,3,16)
    const float* w1 = (const float*)d_in[2];   // (64,64,3,3)
    const float* w2 = (const float*)d_in[3];   // (64,64,3,3)
    float* outp     = (float*)d_out;           // (8,32,32,64)

    half_t* wsB  = (half_t*)d_ws;              // 589824 halves (~1.13 MB)
    half_t* wsW2 = wsB + WSB_HALVES;           // 36864 halves

    kan_prep_wcp<<<(WSB_HALVES + 255) / 256, 256, 0, stream>>>(cp, w1, wsB);
    kan_prep_w2 <<<(WSW2_HALVES + 255) / 256, 256, 0, stream>>>(w2, wsW2);
    kan_conv_wmma<<<NPIX / 64, 128, 0, stream>>>(x, wsB, wsW2, outp);
}
